// HpoolGNN_60361470378388
// MI455X (gfx1250) — matrix operand order repaired and checked by reading, not compiled, not run on hardware
//
#include <hip/hip_runtime.h>
#include <cstdint>
#include <cstddef>

typedef __attribute__((ext_vector_type(16))) _Float16 v16h;
typedef __attribute__((ext_vector_type(8)))  float    v8f;
typedef __attribute__((ext_vector_type(4)))  unsigned int v4u;
typedef __attribute__((ext_vector_type(8)))  int      v8i;
typedef __attribute__((ext_vector_type(4)))  int      v4i;

#define BN_INV 0.9995003747f /* 1/sqrt(1+1e-3) */

// ---------------------------------------------------------------- converts
__global__ void k_cvt_f16(const float* __restrict__ src, _Float16* __restrict__ dst, long n) {
    long i = (long)blockIdx.x * blockDim.x + threadIdx.x;
    long stride = (long)gridDim.x * blockDim.x;
    for (; i < n; i += stride) dst[i] = (_Float16)src[i];
}

// transpose-convert weight: src f32 [K][N] row-major -> dst f16 [N][K]
__global__ void k_wt(const float* __restrict__ src, _Float16* __restrict__ dst, int K, int N) {
    int t = blockIdx.x * blockDim.x + threadIdx.x;
    if (t < K * N) {
        int k = t / N, n = t - k * N;
        dst[(size_t)n * K + k] = (_Float16)src[t];
    }
}

// ------------------------------------------------- TDM descriptor + issue
// 2D tile load: global [rows][K] panel (row stride = ld elements, f16) -> LDS,
// padded with 4 dwords (8 halfs) after every row so fragment rows skew across banks.
__device__ __forceinline__ void tdm_load_2d(unsigned lds_addr,
                                            unsigned long long gaddr,
                                            unsigned K, unsigned rows,
                                            unsigned ld, unsigned padItv)
{
    v4u g0;
    g0.x = 1u;                                   // count=1, user descriptor
    g0.y = lds_addr;                             // LDS byte address
    g0.z = (unsigned)(gaddr & 0xFFFFFFFFull);    // global_addr[31:0]
    g0.w = (unsigned)((gaddr >> 32) & 0x1FFFFFFull) | (2u << 30); // addr[56:32] | type=2

    unsigned long long stride0 = (unsigned long long)ld; // in data_size units
    v8i g1;
    g1[0] = (int)((1u << 16)                     // data_size = 1 (2 bytes)
                | (1u << 20)                     // pad_enable
                | (padItv << 22)                 // pad_interval: (1<<code)*8B = row bytes
                | (3u << 25));                   // pad_amount code 3 = 4 dwords
    g1[1] = (int)((K & 0xFFFFu) << 16);          // tensor_dim0[15:0]
    g1[2] = (int)((K >> 16) | ((rows & 0xFFFFu) << 16)); // dim0 hi | tensor_dim1 lo
    g1[3] = (int)((rows >> 16) | (K << 16));     // dim1 hi | tile_dim0 = K
    g1[4] = (int)rows;                           // tile_dim1 = rows, tile_dim2 = 0
    g1[5] = (int)(stride0 & 0xFFFFFFFFull);      // tensor_dim0_stride lo
    g1[6] = (int)((stride0 >> 32) & 0xFFFFull);  // stride hi (dim1 stride unused, 2D)
    g1[7] = 0;

    v4i z4 = {0, 0, 0, 0};
#if __has_include(<hip/amd_detail/amd_gfx1250_TDM.h>)
    v8i z8 = {0, 0, 0, 0, 0, 0, 0, 0};
    __builtin_amdgcn_tensor_load_to_lds(g0, g1, z4, z4, z8, 0);
#else
    __builtin_amdgcn_tensor_load_to_lds(g0, g1, z4, z4, 0);
#endif
}

// ---------------------------------------------------------------- WMMA GEMM
// D = epilogue(A[M,K] @ B^T[N,K] + C).  A row-major [M][K] f16, B stored [N][K] f16.
// mode 0: raw, store D transposed: D[n*ldd + m]      (produces next [N][K] operand)
// mode 1: tanh(((acc+bias)*BN_INV)*gain+beta), store D[m*ldd + n]
// mode 2: tanh(acc+bias),                      store D[m*ldd + n]
// Block: 256 threads = 8 waves, block tile 64(M) x 32(N), wave tile 16x16.
// A/B panels staged in LDS by the Tensor Data Mover (wave 0 issues, TENSORcnt wait).
union Frag16 { v16h v; uint4 q[2]; };

#define LDSA_MAX (64 * 264)   /* 64 rows x (256+8) halfs */
#define LDSB_MAX (32 * 264)

__global__ void __launch_bounds__(256)
k_wmma_gemm(const _Float16* __restrict__ A, const _Float16* __restrict__ B,
            _Float16* __restrict__ D,
            int K, int lda, int ldb, int ldd,
            long sA_, long sB_, long sD_,
            const float* __restrict__ bias, const float* __restrict__ gain,
            const float* __restrict__ beta, int mode, int padItv)
{
    __shared__ __align__(16) _Float16 sA[LDSA_MAX];
    __shared__ __align__(16) _Float16 sB[LDSB_MAX];

    const int lane = threadIdx.x & 31;
    const int wave = threadIdx.x >> 5;
    const int m0 = blockIdx.x * 64;
    const int n0 = blockIdx.y * 32;
    A += (long)blockIdx.z * sA_;
    B += (long)blockIdx.z * sB_;
    D += (long)blockIdx.z * sD_;

    // --- stage panels into LDS via TDM (one wave issues; EXEC-uniform branch) ---
    if (threadIdx.x < 32) {
        tdm_load_2d((unsigned)(uintptr_t)&sA[0],
                    (unsigned long long)(uintptr_t)(A + (size_t)m0 * lda),
                    (unsigned)K, 64u, (unsigned)lda, (unsigned)padItv);
        tdm_load_2d((unsigned)(uintptr_t)&sB[0],
                    (unsigned long long)(uintptr_t)(B + (size_t)n0 * ldb),
                    (unsigned)K, 32u, (unsigned)ldb, (unsigned)padItv);
        __builtin_amdgcn_s_wait_tensorcnt(0);
    }
    __syncthreads();

    const int ldsRow = K + 8;   // row stride in halfs after TDM padding (4 dwords/row)

    // A-operand layout: lanes 0-15 row M=lane, K halves {0..15};
    // lanes 16-31 row M=lane-16, K halves {16..31}.
    const int arowL = (wave >> 1) * 16 + (lane & 15);
    const int akb   = (lane >> 4) * 16;
    const int bcolL = (wave & 1) * 16 + (lane & 15);
    const int bkb   = (lane >> 4) * 16;

    const _Float16* pA = sA + (size_t)arowL * ldsRow + akb;
    const _Float16* pB = sB + (size_t)bcolL * ldsRow + bkb;

    v8f c = {};
    for (int k0 = 0; k0 < K; k0 += 64) {   // 2x unrolled (K is a multiple of 64)
        Frag16 fa0, fb0, fa1, fb1;
        fa0.q[0] = *reinterpret_cast<const uint4*>(pA + k0);
        fa0.q[1] = *reinterpret_cast<const uint4*>(pA + k0 + 8);
        fb0.q[0] = *reinterpret_cast<const uint4*>(pB + k0);
        fb0.q[1] = *reinterpret_cast<const uint4*>(pB + k0 + 8);
        fa1.q[0] = *reinterpret_cast<const uint4*>(pA + k0 + 32);
        fa1.q[1] = *reinterpret_cast<const uint4*>(pA + k0 + 40);
        fb1.q[0] = *reinterpret_cast<const uint4*>(pB + k0 + 32);
        fb1.q[1] = *reinterpret_cast<const uint4*>(pB + k0 + 40);
        c = __builtin_amdgcn_wmma_f32_16x16x32_f16(false, fa0.v, false, fb0.v,
                                                   (short)0, c, false, false);
        c = __builtin_amdgcn_wmma_f32_16x16x32_f16(false, fa1.v, false, fb1.v,
                                                   (short)0, c, false, false);
    }

    float cc[8];
#pragma unroll
    for (int r = 0; r < 8; ++r) cc[r] = c[r];

    // C/D layout: VGPR r -> (M = r + 8*(lane>>4), N = lane&15)
    const int mbase = m0 + (wave >> 1) * 16 + ((lane >> 4) * 8);
    const int n = n0 + (wave & 1) * 16 + (lane & 15);
    if (mode == 0) {
        _Float16* dp = D + (size_t)n * ldd + mbase;
#pragma unroll
        for (int r = 0; r < 8; ++r) dp[r] = (_Float16)cc[r];
    } else if (mode == 1) {
        const float bb = bias[n], gg = gain[n], be = beta[n];
#pragma unroll
        for (int r = 0; r < 8; ++r)
            D[(size_t)(mbase + r) * ldd + n] =
                (_Float16)tanhf(((cc[r] + bb) * BN_INV) * gg + be);
    } else {
        const float bb = bias[n];
#pragma unroll
        for (int r = 0; r < 8; ++r)
            D[(size_t)(mbase + r) * ldd + n] = (_Float16)tanhf(cc[r] + bb);
    }
}

// ---------------------------------------------------------------- TopK (per graph)
__global__ void __launch_bounds__(256)
k_topk(const _Float16* __restrict__ h, const float* __restrict__ p,
       float* __restrict__ ty, int* __restrict__ idx)
{
    __shared__ float sy[256];
    const int g = blockIdx.x, n = threadIdx.x;
    const _Float16* hg = h + (size_t)g * 65536 + (size_t)n * 256;
    float pn2 = 0.f, acc = 0.f;
    for (int k = 0; k < 256; ++k) {
        float pv = p[k];
        pn2 += pv * pv;
        acc += (float)hg[k] * pv;
    }
    const float yn = acc * rsqrtf(pn2);
    sy[n] = yn;
    __syncthreads();
    int cnt = 0;
    for (int j = 0; j < 256; ++j) {
        float yj = sy[j];
        cnt += (yj > yn) || (yj == yn && j < n);
    }
    ty[g * 256 + n] = tanhf(yn);
    if (cnt < 128) idx[g * 128 + cnt] = n;
}

// gather pooled h (scaled by tanh(y)) and pooled adjacency
__global__ void __launch_bounds__(256)
k_gather(const _Float16* __restrict__ h, const _Float16* __restrict__ aH,
         const float* __restrict__ ty, const int* __restrict__ idx,
         _Float16* __restrict__ hP, _Float16* __restrict__ aP)
{
    __shared__ int   sI[128];
    __shared__ float sT[128];
    const int g = blockIdx.x, tid = threadIdx.x;
    if (tid < 128) {
        int v = idx[g * 128 + tid];
        sI[tid] = v;
        sT[tid] = ty[g * 256 + v];
    }
    __syncthreads();
    const size_t hg = (size_t)g * 65536;
    for (int e = tid; e < 128 * 256; e += 256) {
        int r = e >> 8, f = e & 255;
        hP[(size_t)g * 32768 + e] = (_Float16)((float)h[hg + (size_t)sI[r] * 256 + f] * sT[r]);
    }
    for (int e = tid; e < 128 * 128; e += 256) {
        int r = e >> 7, cix = e & 127;
        aP[(size_t)g * 16384 + e] = aH[hg + (size_t)sI[r] * 256 + sI[cix]];
    }
}

// global sum pool over 128 kept nodes
__global__ void __launch_bounds__(256)
k_pool(const _Float16* __restrict__ hP, _Float16* __restrict__ pooled)
{
    const int g = blockIdx.x, f = threadIdx.x;
    float s = 0.f;
    const _Float16* hg = hP + (size_t)g * 32768 + f;
    for (int nn = 0; nn < 128; ++nn) s += (float)hg[(size_t)nn * 256];
    pooled[g * 256 + f] = (_Float16)s;
}

// final dense [512,256]@[256,10] + BN, no activation, f32 out
__global__ void __launch_bounds__(256)
k_post1(const _Float16* __restrict__ h, const float* __restrict__ W,
        const float* __restrict__ b, const float* __restrict__ gn,
        const float* __restrict__ be, float* __restrict__ out)
{
    int t = blockIdx.x * blockDim.x + threadIdx.x;
    if (t >= 512 * 10) return;
    int g = t / 10, o = t - g * 10;
    float acc = 0.f;
    const _Float16* hg = h + (size_t)g * 256;
    for (int k = 0; k < 256; ++k) acc += (float)hg[k] * W[k * 10 + o];
    out[t] = ((acc + b[o]) * BN_INV) * gn[o] + be[o];
}

// ---------------------------------------------------------------- launch
extern "C" void kernel_launch(void* const* d_in, const int* /*in_sizes*/, int /*n_in*/,
                              void* d_out, int /*out_size*/, void* d_ws, size_t /*ws_size*/,
                              hipStream_t stream)
{
    const float* x     = (const float*)d_in[0];
    const float* a     = (const float*)d_in[1];
    const float* preW[3]  = {(const float*)d_in[2], (const float*)d_in[6],  (const float*)d_in[10]};
    const float* preB[3]  = {(const float*)d_in[3], (const float*)d_in[7],  (const float*)d_in[11]};
    const float* preG[3]  = {(const float*)d_in[4], (const float*)d_in[8],  (const float*)d_in[12]};
    const float* preBe[3] = {(const float*)d_in[5], (const float*)d_in[9],  (const float*)d_in[13]};
    const float* gcnW  = (const float*)d_in[14];
    const float* gcnB  = (const float*)d_in[15];
    const float* topkP = (const float*)d_in[16];
    const float* postW0 = (const float*)d_in[17];
    const float* postB0 = (const float*)d_in[18];
    const float* postG0 = (const float*)d_in[19];
    const float* postBe0= (const float*)d_in[20];
    const float* postW1 = (const float*)d_in[21];
    const float* postB1 = (const float*)d_in[22];
    const float* postG1 = (const float*)d_in[23];
    const float* postBe1= (const float*)d_in[24];

    char* ws = (char*)d_ws;
    size_t off = 0;
    auto alloc = [&](size_t bytes) { size_t c = off; off += (bytes + 255) & ~(size_t)255; return c; };

    _Float16* xh  = (_Float16*)(ws + alloc(8388608ull  * 2)); // x in f16  [G*N,64]
    _Float16* hA  = (_Float16*)(ws + alloc(33554432ull * 2)); // activations [G,256,256]
    _Float16* hB  = (_Float16*)(ws + alloc(33554432ull * 2)); // (h·W)^T scratch
    _Float16* aH  = (_Float16*)(ws + alloc(33554432ull * 2)); // adjacency f16
    _Float16* aP  = (_Float16*)(ws + alloc(8388608ull  * 2)); // pooled adjacency [G,128,128]
    _Float16* hP  = (_Float16*)(ws + alloc(16777216ull * 2)); // pooled h [G,128,256]
    _Float16* w0t = (_Float16*)(ws + alloc(16384ull  * 2));
    _Float16* w1t = (_Float16*)(ws + alloc(65536ull  * 2));
    _Float16* w2t = (_Float16*)(ws + alloc(65536ull  * 2));
    _Float16* gWt = (_Float16*)(ws + alloc(327680ull * 2));   // 5 x [256][256]
    _Float16* pW0t= (_Float16*)(ws + alloc(65536ull  * 2));
    float*    ty  = (float*)   (ws + alloc(131072ull * 4));   // tanh(scores)
    int*      idx = (int*)     (ws + alloc(65536ull  * 4));   // [G,128]
    _Float16* pooled = (_Float16*)(ws + alloc(131072ull * 2));
    _Float16* post0h = (_Float16*)(ws + alloc(131072ull * 2));

    // pad_interval code: (1<<code) 8-byte units per row = K*2 bytes -> code = log2(K)-2
    const int PI64 = 4, PI128 = 5, PI256 = 6;

    // --- convert inputs / weights to f16 (weights transposed to [N][K]) ---
    k_cvt_f16<<<4096, 256, 0, stream>>>(x, xh, 8388608L);
    k_cvt_f16<<<8192, 256, 0, stream>>>(a, aH, 33554432L);
    k_wt<<<64,  256, 0, stream>>>(preW[0], w0t, 64, 256);
    k_wt<<<256, 256, 0, stream>>>(preW[1], w1t, 256, 256);
    k_wt<<<256, 256, 0, stream>>>(preW[2], w2t, 256, 256);
    for (int l = 0; l < 5; ++l)
        k_wt<<<256, 256, 0, stream>>>(gcnW + (size_t)l * 65536, gWt + (size_t)l * 65536, 256, 256);
    k_wt<<<256, 256, 0, stream>>>(postW0, pW0t, 256, 256);

    // --- pre-MLP: three dense+BN+tanh layers (single big GEMM each) ---
    k_wmma_gemm<<<dim3(2048, 8, 1), 256, 0, stream>>>(xh, w0t, hA, 64, 64, 64, 256,
        0, 0, 0, preB[0], preG[0], preBe[0], 1, PI64);
    k_wmma_gemm<<<dim3(2048, 8, 1), 256, 0, stream>>>(hA, w1t, hB, 256, 256, 256, 256,
        0, 0, 0, preB[1], preG[1], preBe[1], 1, PI256);
    k_wmma_gemm<<<dim3(2048, 8, 1), 256, 0, stream>>>(hB, w2t, hA, 256, 256, 256, 256,
        0, 0, 0, preB[2], preG[2], preBe[2], 1, PI256);

    // --- GCN layers 0..2 (full graphs, batched over G=512) ---
    for (int l = 0; l < 3; ++l) {
        // T^T = (h @ W_l)^T   -> hB laid out [feat 256][node 256] per graph
        k_wmma_gemm<<<dim3(4, 8, 512), 256, 0, stream>>>(hA, gWt + (size_t)l * 65536, hB,
            256, 256, 256, 256, 65536, 0, 65536, nullptr, nullptr, nullptr, 0, PI256);
        // h' = tanh(A_hat @ T + b_l)
        k_wmma_gemm<<<dim3(4, 8, 512), 256, 0, stream>>>(aH, hB, hA,
            256, 256, 256, 256, 65536, 65536, 65536,
            gcnB + (size_t)l * 256, nullptr, nullptr, 2, PI256);
    }

    // --- TopK pooling (ratio 0.5) + gather ---
    k_topk<<<512, 256, 0, stream>>>(hA, topkP, ty, idx);
    k_gather<<<512, 256, 0, stream>>>(hA, aH, ty, idx, hP, aP);

    // --- GCN layers 3..4 on pooled graphs (128 nodes) ---
    for (int l = 3; l < 5; ++l) {
        k_wmma_gemm<<<dim3(2, 8, 512), 256, 0, stream>>>(hP, gWt + (size_t)l * 65536, hB,
            256, 256, 256, 128, 32768, 0, 32768, nullptr, nullptr, nullptr, 0, PI256);
        k_wmma_gemm<<<dim3(2, 8, 512), 256, 0, stream>>>(aP, hB, hP,
            128, 128, 128, 256, 16384, 32768, 32768,
            gcnB + (size_t)l * 256, nullptr, nullptr, 2, PI128);
    }

    // --- global sum pool + post-MLP ---
    k_pool<<<512, 256, 0, stream>>>(hP, pooled);
    k_wmma_gemm<<<dim3(8, 8, 1), 256, 0, stream>>>(pooled, pW0t, post0h, 256, 256, 256, 256,
        0, 0, 0, postB0, postG0, postBe0, 1, PI256);
    k_post1<<<20, 256, 0, stream>>>(post0h, postW1, postB1, postG1, postBe1, (float*)d_out);
}
